// ShiftSpecificCrossAttention_38293928411607
// MI455X (gfx1250) — matrix operand order repaired and checked by reading, not compiled, not run on hardware
//
#include <hip/hip_runtime.h>

// ShiftSpecificCrossAttention fused CDNA5 (gfx1250) implementation.
//   prep_weights_kernel : Wk/Wv/Wo fp32 -> bf16 once into d_ws (hot in 192MB L2)
//   prep_q_kernel       : Q = shift_embed @ Wq^T + bq -> bf16 [8,512] in d_ws
//   fused_attn_kernel   : 1 block per batch row b (2048 blocks, 256 thr = 8 waves)
//       phase1: stage X=neighbor_enc[b] (64x512) fp32->bf16 into swizzled LDS
//       phase2: Kp = X @ Wk^T + bk   (v_wmma_f32_16x16x32_bf16, f32 accum) -> LDS
//       phase3: logits via 16B v8bf LDS loads + dist bias + mask + wave32 softmax
//       phase4: Vp = X @ Wv^T + bv   (WMMA, overwrites Kp LDS region)
//       phase5: out = attn @ Vp (vectorized b128 LDS reads) -> bf16 LDS
//       phase6: y = out @ Wo^T + bo + shift_embed (WMMA, M=8 padded to 16)
//       phase7: LayerNorm + any_valid mask -> d_out
// LDS = exactly 160KB/block -> 2 blocks per 320KB WGP.

#define D   512
#define S   8
#define H   8
#define HD  64
#define KN  64

typedef __attribute__((ext_vector_type(16))) __bf16 v16bf;
typedef __attribute__((ext_vector_type(8)))  __bf16 v8bf;
typedef __attribute__((ext_vector_type(4)))  __bf16 v4bf;
typedef __attribute__((ext_vector_type(8)))  float  v8f;

// XOR swizzle for 1024B rows: XOR byteCol bits[7:4] with row[3:0] so lanes
// walking different rows at the same column hit different banks.
__device__ __forceinline__ int swz(int row, int byteCol) {
  return (row << 10) + (byteCol ^ ((row & 15) << 4));
}

// XOR swizzle for the attn[64 rows][64 floats] array (256B rows): spread the
// 4 s-rows read simultaneously in phase 5 across different 64B bank groups.
__device__ __forceinline__ int aswz(int row, int k) {
  return row * KN + (k ^ ((row & 3) << 4));
}

// Load A-fragment (16x32 bf16) in this file's contiguous-k wave32 layout:
// lane[3:0]=M row; lane[4] selects K 0..15 vs 16..31 (contiguous, matches B).
// Caller passes kElem = kBase + (lane[4] ? 16 : 0); we fetch [kElem, kElem+8)
// and [kElem+8, kElem+16) as two 16B LDS loads.
__device__ __forceinline__ v16bf ldA(const char* base, int row, int kElem) {
  v8bf lo = *(const v8bf*)(base + swz(row, kElem * 2));
  v8bf hi = *(const v8bf*)(base + swz(row, (kElem + 8) * 2));
  v16bf a;
#pragma unroll
  for (int j = 0; j < 8; ++j) { a[j] = lo[j]; a[j + 8] = hi[j]; }
  return a;
}

// One [64,512] x [512,512]^T projection: KV = X @ W^T + bias, bf16 result into
// swizzled LDS. Each wave owns 4 N-tiles of 16 cols; loops all 4 M-tiles with
// the B fragment held in registers (reused 4x per load).
__device__ __forceinline__ void proj(const __bf16* __restrict__ W,
                                     const float* __restrict__ bias,
                                     const char* Xs, char* KVs,
                                     int wave, int l15, int mhalf) {
#pragma unroll
  for (int i = 0; i < 4; ++i) {
    const int col = (wave * 4 + i) * 16 + l15;        // output column == W row
    const __bf16* wrow = W + (size_t)col * D;
    v8f acc0 = {}, acc1 = {}, acc2 = {}, acc3 = {};
    for (int ks = 0; ks < 16; ++ks) {
      const int kb = ks * 32;
      // B 32x16: lane[3:0]=N col, lane[4] selects K 0..15 vs 16..31; B column
      // col is row `col` of W (contiguous) -> one 32B global load per lane.
      v16bf B = *(const v16bf*)(wrow + kb + mhalf * 16);
      v16bf A0 = ldA(Xs,  0 + l15, kb + mhalf * 16);
      v16bf A1 = ldA(Xs, 16 + l15, kb + mhalf * 16);
      v16bf A2 = ldA(Xs, 32 + l15, kb + mhalf * 16);
      v16bf A3 = ldA(Xs, 48 + l15, kb + mhalf * 16);
      acc0 = __builtin_amdgcn_wmma_f32_16x16x32_bf16(false, A0, false, B, (short)0, acc0, false, false);
      acc1 = __builtin_amdgcn_wmma_f32_16x16x32_bf16(false, A1, false, B, (short)0, acc1, false, false);
      acc2 = __builtin_amdgcn_wmma_f32_16x16x32_bf16(false, A2, false, B, (short)0, acc2, false, false);
      acc3 = __builtin_amdgcn_wmma_f32_16x16x32_bf16(false, A3, false, B, (short)0, acc3, false, false);
    }
    const float bc = bias[col];
    // C/D layout: VGPR j -> M=j (lanes 0-15) / M=j+8 (lanes 16-31), N=lane[3:0]
#pragma unroll
    for (int j = 0; j < 8; ++j) {
      const int r = j + mhalf * 8;
      *(__bf16*)(KVs + swz( 0 + r, col * 2)) = (__bf16)(acc0[j] + bc);
      *(__bf16*)(KVs + swz(16 + r, col * 2)) = (__bf16)(acc1[j] + bc);
      *(__bf16*)(KVs + swz(32 + r, col * 2)) = (__bf16)(acc2[j] + bc);
      *(__bf16*)(KVs + swz(48 + r, col * 2)) = (__bf16)(acc3[j] + bc);
    }
  }
}

__global__ void __launch_bounds__(256)
prep_weights_kernel(const float* __restrict__ Wk, const float* __restrict__ Wv,
                    const float* __restrict__ Wo,
                    __bf16* __restrict__ wk, __bf16* __restrict__ wv,
                    __bf16* __restrict__ wo) {
  int i = blockIdx.x * 256 + threadIdx.x;
  if (i < D * D) {
    wk[i] = (__bf16)Wk[i];
    wv[i] = (__bf16)Wv[i];
    wo[i] = (__bf16)Wo[i];
  }
}

__global__ void __launch_bounds__(256)
prep_q_kernel(const float* __restrict__ se, const float* __restrict__ Wq,
              const float* __restrict__ bq, __bf16* __restrict__ q) {
  int i = blockIdx.x * 256 + threadIdx.x;  // S*D = 4096 outputs
  int s = i >> 9, c = i & (D - 1);
  const float* sr = se + s * D;
  const float* wr = Wq + (size_t)c * D;
  float acc = bq[c];
  for (int d = 0; d < D; ++d) acc = __builtin_fmaf(sr[d], wr[d], acc);
  q[i] = (__bf16)acc;
}

__global__ void __launch_bounds__(256)
fused_attn_kernel(const float* __restrict__ nbr,            // [B,KN,D] fp32
                  const unsigned char* __restrict__ valid,  // [B,KN] bool (1B)
                  const float* __restrict__ dist,           // [B,KN]
                  const float* __restrict__ se,             // [S,D]
                  const float* __restrict__ bk, const float* __restrict__ bv,
                  const float* __restrict__ bo,
                  const float* __restrict__ lnw, const float* __restrict__ lnb,
                  const float* __restrict__ cscale,         // [1]
                  const __bf16* __restrict__ wk, const __bf16* __restrict__ wv,
                  const __bf16* __restrict__ wo,
                  const __bf16* __restrict__ qg,            // [S*D] bf16
                  float* __restrict__ out) {                // [B,S,D]
  __shared__ char smem[163840];                 // 160KB: 2 blocks per WGP
  char*   Xs   = smem;                          // 64KB  bf16 X, swizzled
  char*   KVs  = smem + 65536;                  // 64KB  bf16 Kp then Vp
  __bf16* Qs   = (__bf16*)(smem + 131072);      //  8KB  Q bf16 [S*D]
  float*  attn = (float*)(smem + 139264);       // 16KB  attn[H*S][KN] / y[S][D]
  char*   outA = smem + 155648;                 //  8KB  bf16 out[S][D], swizzled

  const int tid   = threadIdx.x;
  const int wave  = tid >> 5;
  const int lane  = tid & 31;
  const int l15   = lane & 15;
  const int mhalf = (lane >> 4) & 1;
  const int b     = blockIdx.x;

  // ---- Phase 1: stage X and Q into LDS -------------------------------------
  {
    const float* xg = nbr + (size_t)b * (KN * D);
#pragma unroll
    for (int it = 0; it < (KN * D / 4) / 256; ++it) {   // 32 float4 per thread
      int g = it * 256 + tid;
      int e = g * 4;
      int row = e >> 9;
      int col = e & (D - 1);
      float4 v = ((const float4*)xg)[g];
      v4bf p;
      p[0] = (__bf16)v.x; p[1] = (__bf16)v.y; p[2] = (__bf16)v.z; p[3] = (__bf16)v.w;
      *(v4bf*)(Xs + swz(row, col * 2)) = p;
    }
#pragma unroll
    for (int it = 0; it < 2; ++it) {                    // 8KB of Q as int4
      int g = it * 256 + tid;
      ((int4*)Qs)[g] = ((const int4*)qg)[g];
    }
  }
  __syncthreads();

  // ---- Phase 2: Kp = X @ Wk^T + bk (WMMA) ----------------------------------
  proj(wk, bk, Xs, KVs, wave, l15, mhalf);
  __syncthreads();

  // ---- Phase 3: logits + bias + mask + softmax (wave32 reductions) ---------
  {
    const int h = wave;                    // one head per wave
    const int k0 = lane, k1 = lane + 32;   // each lane owns 2 neighbors
    const float csc = cscale[0];
    const float scale = 0.125f;            // HD^-0.5
    const bool  va0 = valid[b * KN + k0] != 0;
    const bool  va1 = valid[b * KN + k1] != 0;
    const float b0 = dist[b * KN + k0] * csc;
    const float b1 = dist[b * KN + k1] * csc;
    for (int s = 0; s < S; ++s) {
      const char* qb = (const char*)(Qs + s * D + h * HD);
      float a0 = 0.f, a1 = 0.f;
#pragma unroll
      for (int c = 0; c < 8; ++c) {        // 8 x 16B chunks cover HD=64
        v8bf qv = *(const v8bf*)(qb + c * 16);                 // LDS broadcast
        v8bf x0 = *(const v8bf*)(KVs + swz(k0, h * 128 + c * 16));
        v8bf x1 = *(const v8bf*)(KVs + swz(k1, h * 128 + c * 16));
#pragma unroll
        for (int j = 0; j < 8; ++j) {
          float qf = (float)qv[j];
          a0 = __builtin_fmaf(qf, (float)x0[j], a0);
          a1 = __builtin_fmaf(qf, (float)x1[j], a1);
        }
      }
      float l0 = va0 ? __builtin_fmaf(a0, scale, b0) : -10000.f;
      float l1 = va1 ? __builtin_fmaf(a1, scale, b1) : -10000.f;
      float m = fmaxf(l0, l1);
#pragma unroll
      for (int off = 16; off >= 1; off >>= 1) m = fmaxf(m, __shfl_xor(m, off));
      float p0 = __expf(l0 - m), p1 = __expf(l1 - m);
      float sum = p0 + p1;
#pragma unroll
      for (int off = 16; off >= 1; off >>= 1) sum += __shfl_xor(sum, off);
      float inv = 1.0f / sum;
      attn[aswz(h * S + s, k0)] = p0 * inv;
      attn[aswz(h * S + s, k1)] = p1 * inv;
    }
  }
  __syncthreads();

  // ---- Phase 4: Vp = X @ Wv^T + bv (WMMA, overwrites Kp) -------------------
  proj(wv, bv, Xs, KVs, wave, l15, mhalf);
  __syncthreads();

  // ---- Phase 5: out = attn @ Vp -> bf16 LDS (vectorized) -------------------
  {
    const int h  = wave;
    const int dg = lane & 7;               // 8-element d-group within the head
    const int sg = lane >> 3;              // 0..3 -> s in {sg, sg+4}
#pragma unroll
    for (int sp = 0; sp < 2; ++sp) {
      const int s = sg + sp * 4;
      float acc[8] = {0.f, 0.f, 0.f, 0.f, 0.f, 0.f, 0.f, 0.f};
#pragma unroll 8
      for (int k = 0; k < KN; ++k) {
        float a = attn[aswz(h * S + s, k)];
        v8bf x = *(const v8bf*)(KVs + swz(k, h * 128 + dg * 16));
#pragma unroll
        for (int j = 0; j < 8; ++j) acc[j] = __builtin_fmaf(a, (float)x[j], acc[j]);
      }
      v8bf o;
#pragma unroll
      for (int j = 0; j < 8; ++j) o[j] = (__bf16)acc[j];
      *(v8bf*)(outA + swz(s, h * 128 + dg * 16)) = o;
    }
  }
  __syncthreads();

  // ---- Phase 6: y = out @ Wo^T + bo + shift_embed (WMMA, M padded to 16) ---
  float* yF = attn;   // reuse attn region as y[S][D] fp32
  {
#pragma unroll
    for (int i = 0; i < 4; ++i) {
      const int col = (wave * 4 + i) * 16 + l15;
      const __bf16* wrow = wo + (size_t)col * D;
      v8f acc = {};
      for (int ks = 0; ks < 16; ++ks) {
        const int kb = ks * 32;
        v16bf B = *(const v16bf*)(wrow + kb + mhalf * 16);
        v16bf A = {};
        if (l15 < 8) A = ldA(outA, l15, kb + mhalf * 16);  // rows 8..15 are zero
        acc = __builtin_amdgcn_wmma_f32_16x16x32_bf16(false, A, false, B, (short)0, acc, false, false);
      }
      if (mhalf == 0) {            // valid rows M=0..7 live in lanes 0-15, VGPR j
        const float bc = bo[col];
#pragma unroll
        for (int j = 0; j < 8; ++j)
          yF[j * D + col] = acc[j] + bc + se[j * D + col];
      }
    }
  }
  __syncthreads();

  // ---- Phase 7: LayerNorm + any_valid mask + store -------------------------
  {
    const int s = wave;            // one shift per wave
    int vv = (valid[b * KN + lane] != 0) || (valid[b * KN + lane + 32] != 0);
    const float anyf = __any(vv) ? 1.f : 0.f;
    float sum = 0.f;
#pragma unroll
    for (int j = 0; j < 16; ++j) sum += yF[s * D + lane + 32 * j];
#pragma unroll
    for (int off = 16; off >= 1; off >>= 1) sum += __shfl_xor(sum, off);
    const float mu = sum * (1.0f / D);
    float vs = 0.f;
#pragma unroll
    for (int j = 0; j < 16; ++j) {
      float t = yF[s * D + lane + 32 * j] - mu;
      vs = __builtin_fmaf(t, t, vs);
    }
#pragma unroll
    for (int off = 16; off >= 1; off >>= 1) vs += __shfl_xor(vs, off);
    const float rstd = rsqrtf(vs * (1.0f / D) + 1e-5f);
    float* og = out + ((size_t)b * S + s) * D;
#pragma unroll
    for (int j = 0; j < 16; ++j) {
      int c = lane + 32 * j;
      float x = yF[s * D + c];
      og[c] = ((x - mu) * rstd * lnw[c] + lnb[c]) * anyf;
    }
  }
}

extern "C" void kernel_launch(void* const* d_in, const int* in_sizes, int n_in,
                              void* d_out, int out_size, void* d_ws, size_t ws_size,
                              hipStream_t stream) {
  (void)n_in; (void)out_size; (void)ws_size;
  const float*         nbr   = (const float*)d_in[0];
  const unsigned char* valid = (const unsigned char*)d_in[1];  // jnp bool = 1 byte
  const float*         dist  = (const float*)d_in[2];
  const float*         se    = (const float*)d_in[3];
  const float*         Wq    = (const float*)d_in[4];
  const float*         bq    = (const float*)d_in[5];
  const float*         Wk    = (const float*)d_in[6];
  const float*         bk    = (const float*)d_in[7];
  const float*         Wv    = (const float*)d_in[8];
  const float*         bv    = (const float*)d_in[9];
  const float*         Wo    = (const float*)d_in[10];
  const float*         bo    = (const float*)d_in[11];
  const float*         lnw   = (const float*)d_in[12];
  const float*         lnb   = (const float*)d_in[13];
  const float*         csc   = (const float*)d_in[14];
  float* out = (float*)d_out;

  // Workspace layout (bf16): Wk | Wv | Wo | Q  = 3*512KB + 8KB
  __bf16* wk = (__bf16*)d_ws;
  __bf16* wv = wk + D * D;
  __bf16* wo = wv + D * D;
  __bf16* qg = wo + D * D;

  const int Bn = in_sizes[0] / (KN * D);   // 2048 batch rows

  prep_weights_kernel<<<(D * D + 255) / 256, 256, 0, stream>>>(Wk, Wv, Wo, wk, wv, wo);
  prep_q_kernel<<<(S * D) / 256, 256, 0, stream>>>(se, Wq, bq, qg);
  fused_attn_kernel<<<Bn, 256, 0, stream>>>(nbr, valid, dist, se, bk, bv, bo,
                                            lnw, lnb, csc, wk, wv, wo, qg, out);
}
